// AutoFormer_46935402610760
// MI455X (gfx1250) — matrix operand order repaired and checked by reading, not compiled, not run on hardware
//
#include <hip/hip_runtime.h>
#include <math.h>

// B=8, L=2048, D=1024, H=16, E=64, top_k = int(3*ln(2048)) = 22
#define TOPK 22
#define PI_F 3.14159265358979323846f
#define SPAD 1040   // padded LDS row stride in halves (1024 + 16 pad)

typedef __attribute__((ext_vector_type(16))) _Float16 v16h;
typedef __attribute__((ext_vector_type(8)))  float    v8f;

// ---------------------------------------------------------------------------
// Projection GEMM for Q and K:  Yt[b, c, l] = (X @ W + bias)[b*2048+l, c]
// X: [16384,1024] fp32, W: [1024,1024] fp32.
// Block = 8 waves = 8 M-tiles sharing one N-tile; the W column panel is
// staged once in LDS as f16 [n][k] so each B fragment is one 32B LDS read.
// ---------------------------------------------------------------------------
__global__ __launch_bounds__(256) void proj_gemm_wmma(
    const float* __restrict__ X, const float* __restrict__ W,
    const float* __restrict__ bias, float* __restrict__ Yt)
{
  __shared__ __attribute__((aligned(32))) _Float16 ldsW[16 * SPAD]; // 33.3 KB

  const int K = 1024;
  int wave  = threadIdx.x >> 5;
  int lane  = threadIdx.x & 31;
  int blkM  = blockIdx.x >> 6;          // 128 M-blocks x 8 tiles
  int tileN = blockIdx.x & 63;          // 64 N-tiles
  int n0    = tileN << 4;

  // Cooperative stage: W[:, n0..n0+15] -> f16, transposed [n][k].
  for (int idx = threadIdx.x; idx < 16384; idx += 256) {
    int n = idx & 15, kk = idx >> 4;                  // coalesced row reads
    ldsW[n * SPAD + kk] = (_Float16)W[((size_t)kk << 10) + n0 + n];
  }
  __syncthreads();

  int tileM = (blkM << 3) + wave;
  int half  = lane >> 4;
  int mr    = (tileM << 4) + (lane & 15);   // A row this lane feeds
  int nc    = n0 + (lane & 15);             // B/D column this lane feeds

  v8f acc = {};
  const float4*    xrow = (const float4*)(X + (size_t)mr * K);
  const _Float16*  bcol = ldsW + (size_t)(lane & 15) * SPAD;

  #pragma unroll 8
  for (int k0 = 0; k0 < K; k0 += 32) {
    // A fragment: elements p -> K = k0 + half*16 + p
    int ka = k0 + (half << 4);
    float4 a0 = xrow[ka >> 2];
    float4 a1 = xrow[(ka >> 2) + 1];
    float4 a2 = xrow[(ka >> 2) + 2];
    float4 a3 = xrow[(ka >> 2) + 3];
    v16h af;
    af[0]  = (_Float16)a0.x; af[1]  = (_Float16)a0.y; af[2]  = (_Float16)a0.z; af[3]  = (_Float16)a0.w;
    af[4]  = (_Float16)a1.x; af[5]  = (_Float16)a1.y; af[6]  = (_Float16)a1.z; af[7]  = (_Float16)a1.w;
    af[8]  = (_Float16)a2.x; af[9]  = (_Float16)a2.y; af[10] = (_Float16)a2.z; af[11] = (_Float16)a2.w;
    af[12] = (_Float16)a3.x; af[13] = (_Float16)a3.y; af[14] = (_Float16)a3.z; af[15] = (_Float16)a3.w;
    // B fragment: lane -> column, elements p -> K = k0 + half*16 + p
    // (32B-aligned: SPAD*2 and (k0+half*16)*2 are multiples of 32)
    v16h bf = *(const v16h*)(bcol + k0 + (half << 4));
    acc = __builtin_amdgcn_wmma_f32_16x16x32_f16(false, af, false, bf,
                                                 (short)0, acc, false, false);
  }

  // D layout: lane holds rows M = tileM*16 + half*8 + i, column nc.
  // Yt index for (m, nc): b = m>>11, l = m&2047 -> (b*1024+nc)*2048 + l
  float bb    = bias[nc];
  int   mbase = (tileM << 4) + (half << 3);
  int   b     = mbase >> 11;
  int   lt    = mbase & 2047;           // 8 consecutive l per lane
  float* yp   = Yt + (((size_t)(b << 10) + nc) << 11) + lt;
  float4* yv  = (float4*)yp;
  yv[0] = make_float4(acc[0] + bb, acc[1] + bb, acc[2] + bb, acc[3] + bb);
  yv[1] = make_float4(acc[4] + bb, acc[5] + bb, acc[6] + bb, acc[7] + bb);
}

// ---------------------------------------------------------------------------
// In-LDS Stockham radix-2 FFT, N=2048, natural order in/out, 256 threads.
// src/dst ping-pong; on return `src` holds the result.
// ---------------------------------------------------------------------------
__device__ inline void stockham11(float2*& src, float2*& dst, int tid)
{
  #pragma unroll 1
  for (int s = 0; s < 11; ++s) {
    int   m    = 1 << s;
    float linv = 1.0f / (float)(1024 >> s);
    for (int t = tid; t < 1024; t += 256) {
      int k = t & (m - 1);
      int j = t >> s;
      float2 a = src[k + (j << s)];
      float2 b = src[k + (j << s) + 1024];
      float ang = -PI_F * (float)j * linv;
      float sw, cw;
      __sincosf(ang, &sw, &cw);
      float dx = a.x - b.x, dy = a.y - b.y;
      int o0 = k + (j << (s + 1));
      dst[o0]     = make_float2(a.x + b.x, a.y + b.y);
      dst[o0 + m] = make_float2(dx * cw - dy * sw, dx * sw + dy * cw);
    }
    __syncthreads();
    float2* t2 = src; src = dst; dst = t2;
  }
}

// ---------------------------------------------------------------------------
// One block per channel ch=(b*16+h)*64+e:
//   z = q + i*k ; Z = FFT(z) ; P = qf*conj(kf) via symmetry ; corr = IFFT(P)
//   write corr to d_out in [B,L,H,E]; fused top-22 + softmax -> W8[ch][8]
// ---------------------------------------------------------------------------
__global__ __launch_bounds__(256) void fft_corr_topk(
    const float* __restrict__ qT, const float* __restrict__ kT,
    float* __restrict__ corr_out, float* __restrict__ W8)
{
  __shared__ float2 bufA[2048];
  __shared__ float2 bufB[2048];
  __shared__ float  candV[32 * TOPK];
  __shared__ int    candI[32 * TOPK];

  int tid = threadIdx.x;
  int ch  = blockIdx.x;           // 0..8191
  int b   = ch >> 10;
  int c   = ch & 1023;            // h*64+e
  const float* q = qT + ((size_t)ch << 11);
  const float* k = kT + ((size_t)ch << 11);

  for (int i = tid; i < 2048; i += 256)
    bufA[i] = make_float2(q[i], k[i]);
  __syncthreads();

  float2* src = bufA; float2* dst = bufB;
  stockham11(src, dst, tid);      // Z in src

  // P[f] = qf * conj(kf);  store conj(P) so corr = Re(FFT(conj P))/N
  for (int f = tid; f < 2048; f += 256) {
    float2 z  = src[f];
    float2 zm = src[(2048 - f) & 2047];
    float qr = 0.5f * (z.x + zm.x), qi = 0.5f * (z.y - zm.y);
    float ar = z.x - zm.x, ai = z.y + zm.y;       // z - conj(zm)
    float kr = 0.5f * ai,  ki = -0.5f * ar;       // kf = (z - conj(zm))/(2i)
    float pr = qr * kr + qi * ki;
    float pi = qi * kr - qr * ki;
    dst[f] = make_float2(pr, -pi);
  }
  __syncthreads();

  { float2* t2 = src; src = dst; dst = t2; }
  stockham11(src, dst, tid);      // FFT(conj P) in src

  float* cr = (float*)dst;        // scratch for real corr
  const float invn = 1.0f / 2048.0f;
  for (int l = tid; l < 2048; l += 256) {
    float v = src[l].x * invn;
    cr[l] = v;
    corr_out[(((size_t)(b << 11) + l) << 10) + c] = v;  // [B,L,H,E]
  }
  __syncthreads();

  // ---- top-22 + softmax + mod-8 histogram (wave 0 only) ----
  if (tid < 32) {
    int lane = tid;
    float tv[TOPK]; int ti[TOPK];
    #pragma unroll
    for (int i = 0; i < TOPK; ++i) { tv[i] = -3.4e38f; ti[i] = 0; }
    for (int j = 0; j < 64; ++j) {
      int   idx = (j << 5) + lane;     // bank-friendly stride
      float v   = cr[idx];
      if (v > tv[TOPK - 1]) {
        tv[TOPK - 1] = v; ti[TOPK - 1] = idx;
        #pragma unroll
        for (int p = TOPK - 1; p > 0; --p) {
          if (tv[p] > tv[p - 1]) {
            float fv = tv[p]; tv[p] = tv[p - 1]; tv[p - 1] = fv;
            int   iv = ti[p]; ti[p] = ti[p - 1]; ti[p - 1] = iv;
          }
        }
      }
    }
    #pragma unroll
    for (int i = 0; i < TOPK; ++i) {
      candV[lane * TOPK + i] = tv[i];
      candI[lane * TOPK + i] = ti[i];
    }

    int p = 0;
    float winv[TOPK]; int wini[TOPK];
    #pragma unroll
    for (int r = 0; r < TOPK; ++r) {
      float cv = (p < TOPK) ? candV[lane * TOPK + p] : -3.4e38f;
      int   ci = (p < TOPK) ? candI[lane * TOPK + p] : 0;
      int   cl = lane;
      #pragma unroll
      for (int off = 16; off; off >>= 1) {
        float ov = __shfl_xor(cv, off, 32);
        int   oi = __shfl_xor(ci, off, 32);
        int   ol = __shfl_xor(cl, off, 32);
        if (ov > cv || (ov == cv && ol < cl)) { cv = ov; ci = oi; cl = ol; }
      }
      if (cl == lane) ++p;
      winv[r] = cv; wini[r] = ci;
    }

    float mx = winv[0];             // rounds come out descending
    float ex[TOPK]; float sum = 0.f;
    #pragma unroll
    for (int r = 0; r < TOPK; ++r) { ex[r] = __expf(winv[r] - mx); sum += ex[r]; }
    float inv = 1.0f / sum;
    if (lane < 8) {
      float acc = 0.f;
      #pragma unroll
      for (int r = 0; r < TOPK; ++r)
        if ((wini[r] & 7) == lane) acc += ex[r];
      W8[((size_t)ch << 3) + lane] = acc * inv;   // softmax mass per delay%8 bin
    }
  }
}

// ---------------------------------------------------------------------------
// 64x1024 GEMM (V projection on first 8 rows/batch, and the output GEMM).
// rows_from_seq=1: rows are l=0..7 of each batch in [B,2048,1024];
// rows_from_seq=0: dense [64,1024] input.
// ---------------------------------------------------------------------------
__global__ __launch_bounds__(256) void small_gemm_64x1024(
    const float* __restrict__ Xrows, int rows_from_seq,
    const float* __restrict__ W, const float* __restrict__ bias,
    float* __restrict__ Y)
{
  int o   = blockIdx.x * 256 + threadIdx.x;  // 65536 outputs
  int cc  = o & 1023;
  int row = o >> 10;                         // b*8+m
  const float* x;
  if (rows_from_seq) {
    int b = row >> 3, m = row & 7;
    x = Xrows + (((size_t)(b << 11) + m) << 10);
  } else {
    x = Xrows + ((size_t)row << 10);
  }
  float acc = bias[cc];
  #pragma unroll 4
  for (int kk = 0; kk < 1024; ++kk)
    acc += x[kk] * W[((size_t)kk << 10) + cc];
  Y[o] = acc;
}

// result8[b, m, c] = sum_r W8[b*1024+c, r] * v8[b, (m+r)%8, c]   (m = l%8)
__global__ __launch_bounds__(256) void mix_result8(
    const float* __restrict__ W8, const float* __restrict__ v8,
    float* __restrict__ r8)
{
  int o   = blockIdx.x * 256 + threadIdx.x;
  int cc  = o & 1023;
  int row = o >> 10;
  int b   = row >> 3, m = row & 7;
  int ch  = (b << 10) + cc;
  float acc = 0.f;
  #pragma unroll
  for (int r = 0; r < 8; ++r)
    acc += W8[((size_t)ch << 3) + r] *
           v8[(((size_t)(b << 3) + ((m + r) & 7)) << 10) + cc];
  r8[o] = acc;
}

// out[b, l, c] = out8[b, l%8, c]  (output is period-8 in l)
__global__ __launch_bounds__(256) void broadcast_out(
    const float* __restrict__ out8, float* __restrict__ out)
{
  size_t i  = ((size_t)blockIdx.x << 8) + threadIdx.x;  // 16,777,216
  int    cc = (int)(i & 1023);
  size_t bl = i >> 10;
  int    l  = (int)(bl & 2047);
  int    b  = (int)(bl >> 11);
  out[i] = out8[(((size_t)(b << 3) + (l & 7)) << 10) + cc];
}

// ---------------------------------------------------------------------------
extern "C" void kernel_launch(void* const* d_in, const int* in_sizes, int n_in,
                              void* d_out, int out_size, void* d_ws, size_t ws_size,
                              hipStream_t stream)
{
  (void)in_sizes; (void)n_in; (void)out_size; (void)ws_size;
  const float* queries = (const float*)d_in[0];
  const float* keys    = (const float*)d_in[1];
  const float* values  = (const float*)d_in[2];
  const float* Wq = (const float*)d_in[3]; const float* bq = (const float*)d_in[4];
  const float* Wk = (const float*)d_in[5]; const float* bk = (const float*)d_in[6];
  const float* Wv = (const float*)d_in[7]; const float* bv = (const float*)d_in[8];
  const float* Wo = (const float*)d_in[9]; const float* bo = (const float*)d_in[10];

  float* out      = (float*)d_out;                 // [B,L,D]
  float* corr_out = out + (size_t)16777216;        // [B,L,H,E]

  float* ws   = (float*)d_ws;
  float* qT   = ws;                                // [B,H,E,L]
  float* kT   = qT + (size_t)16777216;             // [B,H,E,L]
  float* v8   = kT + (size_t)16777216;             // [B,8,D]
  float* W8   = v8 + 65536;                        // [B*H*E, 8]
  float* r8   = W8 + 65536;                        // [B,8,D]
  float* out8 = r8 + 65536;                        // [B,8,D]

  proj_gemm_wmma<<<8192, 256, 0, stream>>>(queries, Wq, bq, qT);
  proj_gemm_wmma<<<8192, 256, 0, stream>>>(keys,    Wk, bk, kT);
  small_gemm_64x1024<<<256, 256, 0, stream>>>(values, 1, Wv, bv, v8);
  fft_corr_topk<<<8192, 256, 0, stream>>>(qT, kT, corr_out, W8);
  mix_result8<<<256, 256, 0, stream>>>(W8, v8, r8);
  small_gemm_64x1024<<<256, 256, 0, stream>>>(r8, 0, Wo, bo, out8);
  broadcast_out<<<65536, 256, 0, stream>>>(out8, out);
}
